// NodeEdgeHGTEncoder_15504831939065
// MI455X (gfx1250) — matrix operand order repaired and checked by reading, not compiled, not run on hardware
//
#include <hip/hip_runtime.h>
#include <hip/hip_bf16.h>
#include <math.h>

// ---------------- problem constants (match reference) ----------------
#define NN     20000
#define RR     4
#define EE     40000
#define CIN    256
#define CEDGE  128
#define HIDC   512
#define OUTC   256
#define NHEAD  8
#define DH     64
#define SCALE_ATT 0.125f   // 1/sqrt(64)

typedef __attribute__((ext_vector_type(16))) _Float16 v16h;
typedef __attribute__((ext_vector_type(8)))  _Float16 v8h;
typedef __attribute__((ext_vector_type(4)))  _Float16 v4h;
typedef __attribute__((ext_vector_type(8)))  float    v8f;

// ---------------- tiled WMMA GEMM core ----------------
// Block: 256 threads = 8 wave32, wave grid 4x2 -> block tile 64x64.
// Each wave computes a 16x32 C strip = 2x v_wmma_f32_16x16x32_f16 per K step
// (A fragment reused). A,B fp32 in memory, staged to LDS as f16:
//   As : row-major, stride 32 halfs (64B)  -> A frag = 2x ds_load_b128
//   BsT: transposed  (col-major of B)      -> B frag = 2x ds_load_b128
// Requirements (all call sites satisfy): K % 32 == 0, N % 64 == 0,
// lda/ldb multiples of 4 floats, 16B-aligned base pointers.
#define TM 64
#define TN 64
#define TK 32

__device__ __forceinline__ void gemm_core(
    const float* A, long lda,
    const float* B, long ldb,
    float*       C, long ldc,
    const float* bias,
    int M, int Nn, int K, int act /*0=none,1=relu,2=gelu*/)
{
    __shared__ __attribute__((aligned(16))) _Float16 As [TM][TK];  // 4 KB
    __shared__ __attribute__((aligned(16))) _Float16 BsT[TN][TK];  // 4 KB

    const int tid  = threadIdx.x;
    const int lane = tid & 31;
    const int wave = tid >> 5;     // 0..7
    const int wm   = wave >> 1;    // 0..3 : 16-row strip
    const int wn   = wave & 1;     // 0..1 : 32-col strip

    const int rowBase = blockIdx.x * TM;
    const int colBase = blockIdx.y * TN;

    v8f acc0 = {0.f,0.f,0.f,0.f,0.f,0.f,0.f,0.f};
    v8f acc1 = {0.f,0.f,0.f,0.f,0.f,0.f,0.f,0.f};

    const int half = lane >> 4;    // 0|1
    const int l16  = lane & 15;
    const int arow = wm * 16 + l16;
    const int kbA  = half * 8;     // A: K {0..7|8..15} then +16
    const int kbB  = half * 8;     // B: K {0..7|8..15} then +16

    for (int k0 = 0; k0 < K; k0 += TK) {
        // ---- stage A tile 64x32 (512 float4, 2 per thread), branch-free ----
        #pragma unroll
        for (int it = 0; it < 2; ++it) {
            int i  = tid + it * 256;       // 0..511
            int r  = i >> 3;               // 8 float4 per row
            int c4 = (i & 7) * 4;
            int gr = rowBase + r;
            int grc = gr < M ? gr : (M - 1);
            float4 v = *(const float4*)(A + (long)grc * lda + k0 + c4);
            float s = gr < M ? 1.f : 0.f;
            v4h h4 = { (_Float16)(v.x * s), (_Float16)(v.y * s),
                       (_Float16)(v.z * s), (_Float16)(v.w * s) };
            *(v4h*)&As[r][c4] = h4;        // ds_store_b64
        }
        // ---- stage B tile 32x64 transposed (512 float4, 2 per thread) ----
        #pragma unroll
        for (int it = 0; it < 2; ++it) {
            int i  = tid + it * 256;       // 0..511
            int kr = i >> 4;               // 16 float4 per 64-col row
            int c4 = (i & 15) * 4;
            float4 v = *(const float4*)(B + (long)(k0 + kr) * ldb + colBase + c4);
            BsT[c4 + 0][kr] = (_Float16)v.x;
            BsT[c4 + 1][kr] = (_Float16)v.y;
            BsT[c4 + 2][kr] = (_Float16)v.z;
            BsT[c4 + 3][kr] = (_Float16)v.w;
        }
        // speculative prefetch of next K tile (global_prefetch_b8)
        if (k0 + TK < K) {
            int gr = rowBase + (tid & 63);
            if (gr < M) __builtin_prefetch(&A[(long)gr * lda + k0 + TK], 0, 1);
        }
        __syncthreads();

        // ---- fragments: contiguous 128-bit LDS loads ----
        v8h a0 = *(const v8h*)&As[arow][kbA];
        v8h a1 = *(const v8h*)&As[arow][kbA + 16];
        v16h af = __builtin_shufflevector(a0, a1,
                     0,1,2,3,4,5,6,7,8,9,10,11,12,13,14,15);
        const int bcol0 = wn * 32 + l16;
        v8h b0 = *(const v8h*)&BsT[bcol0][kbB];
        v8h b1 = *(const v8h*)&BsT[bcol0][kbB + 16];
        v16h bf0 = __builtin_shufflevector(b0, b1,
                     0,1,2,3,4,5,6,7,8,9,10,11,12,13,14,15);
        v8h b2 = *(const v8h*)&BsT[bcol0 + 16][kbB];
        v8h b3 = *(const v8h*)&BsT[bcol0 + 16][kbB + 16];
        v16h bf1 = __builtin_shufflevector(b2, b3,
                     0,1,2,3,4,5,6,7,8,9,10,11,12,13,14,15);

        acc0 = __builtin_amdgcn_wmma_f32_16x16x32_f16(
                   false, af, false, bf0, (short)0, acc0, false, false);
        acc1 = __builtin_amdgcn_wmma_f32_16x16x32_f16(
                   false, af, false, bf1, (short)0, acc1, false, false);
        __syncthreads();
    }

    // ---- epilogue: bias + activation + store ----
    const int col0 = colBase + wn * 32 + l16;
    const int col1 = col0 + 16;
    float bv0 = bias ? bias[col0] : 0.f;
    float bv1 = bias ? bias[col1] : 0.f;
    #pragma unroll
    for (int j = 0; j < 8; ++j) {
        int row = rowBase + wm * 16 + j + half * 8;
        if (row < M) {
            float v0 = acc0[j] + bv0;
            float v1 = acc1[j] + bv1;
            if (act == 1) {
                v0 = v0 > 0.f ? v0 : 0.f;
                v1 = v1 > 0.f ? v1 : 0.f;
            } else if (act == 2) {
                v0 = 0.5f * v0 * (1.f + erff(v0 * 0.70710678118f));
                v1 = 0.5f * v1 * (1.f + erff(v1 * 0.70710678118f));
            }
            C[(long)row * ldc + col0] = v0;
            C[(long)row * ldc + col1] = v1;
        }
    }
}

__global__ void __launch_bounds__(256)
gemm_kernel(const float* A, long lda,
            const float* B, long ldb,
            float* C, long ldc,
            const float* bias,
            int M, int Nn, int K, int act)
{
    gemm_core(A, lda, B, ldb, C, ldc, bias, M, Nn, K, act);
}

// per-(r,h) relation transform: rel[r][:, h*64:(h+1)*64] = kqv[:, kOff+h*64:...] @ W[r,h]
__global__ void __launch_bounds__(256)
gemm_rel_kernel(const float* kqv, int kOff, const float* W, float* rel)
{
    int z = blockIdx.z;          // r*NHEAD + h
    int r = z >> 3, h = z & 7;
    gemm_core(kqv + kOff + h * DH, 3 * HIDC,
              W + (long)z * DH * DH, DH,
              rel + (long)r * NN * HIDC + h * DH, HIDC,
              nullptr, NN, DH, DH, 0);
}

// ---------------- attention (gather / softmax / scatter) ----------------
__device__ __forceinline__ void atomicMaxFloat(float* addr, float val)
{
    if (val >= 0.f) atomicMax((int*)addr, __float_as_int(val));
    else            atomicMin((unsigned int*)addr, (unsigned int)__float_as_int(val));
}

__global__ void init_attn(float* amax, float* denom, float* agg)
{
    long i = blockIdx.x * (long)blockDim.x + threadIdx.x;
    if (i < (long)NN * NHEAD) { amax[i] = -__builtin_inff(); denom[i] = 0.f; }
    if (i < (long)NN * HIDC)  agg[i] = 0.f;
}

// one wave per edge: 8 head dot-products of length 64 via butterfly reductions
__global__ void attn_alpha(const float* kqv, const float* krel,
                           const int* eidx, const float* prel,
                           float* alpha, float* amax)
{
    int m    = (blockIdx.x * blockDim.x + threadIdx.x) >> 5;
    int lane = threadIdx.x & 31;
    if (m >= RR * EE) return;
    int r = m / EE, e = m - r * EE;
    int src = eidx[(long)r * 2 * EE + e];
    int dst = eidx[(long)r * 2 * EE + EE + e];
    const float* q = kqv  + (long)dst * (3 * HIDC) + HIDC;          // q slice
    const float* k = krel + ((long)r * NN + src) * HIDC;
    float hs[NHEAD];
    #pragma unroll
    for (int h = 0; h < NHEAD; ++h) hs[h] = 0.f;
    #pragma unroll
    for (int j = 0; j < 16; ++j) {           // head = j>>1 (64 dims = 2 lane-strides)
        int idx = lane + 32 * j;
        hs[j >> 1] += q[idx] * k[idx];
    }
    #pragma unroll
    for (int h = 0; h < NHEAD; ++h) {
        float v = hs[h];
        #pragma unroll
        for (int off = 16; off; off >>= 1) v += __shfl_xor(v, off, 32);
        hs[h] = v;
    }
    if (lane < NHEAD) {
        float a = hs[lane] * prel[r * NHEAD + lane] * SCALE_ATT;
        alpha[(long)m * NHEAD + lane] = a;
        atomicMaxFloat(amax + (long)dst * NHEAD + lane, a);
    }
}

__global__ void attn_exp(float* alpha_ae, const int* eidx,
                         const float* amax, float* denom)
{
    long i = blockIdx.x * (long)blockDim.x + threadIdx.x;
    if (i >= (long)RR * EE * NHEAD) return;
    long m = i >> 3; int h = (int)(i & 7);
    int r = (int)(m / EE), e = (int)(m - (long)r * EE);
    int dst = eidx[(long)r * 2 * EE + EE + e];
    float a = __expf(alpha_ae[i] - amax[(long)dst * NHEAD + h]);
    alpha_ae[i] = a;                               // in-place: alpha -> ae
    atomicAdd(denom + (long)dst * NHEAD + h, a);
}

// one wave per edge: weight v_e by softmax w and scatter-add into agg[dst]
__global__ void attn_scatter(const float* ae, const float* denom,
                             const float* vrel, const int* eidx, float* agg)
{
    int m    = (blockIdx.x * blockDim.x + threadIdx.x) >> 5;
    int lane = threadIdx.x & 31;
    if (m >= RR * EE) return;
    int r = m / EE, e = m - r * EE;
    int src = eidx[(long)r * 2 * EE + e];
    int dst = eidx[(long)r * 2 * EE + EE + e];
    const float* v = vrel + ((long)r * NN + src) * HIDC;
    float w[NHEAD];
    #pragma unroll
    for (int h = 0; h < NHEAD; ++h)
        w[h] = ae[(long)m * NHEAD + h] / (denom[(long)dst * NHEAD + h] + 1e-16f);
    float* out = agg + (long)dst * HIDC;
    #pragma unroll
    for (int j = 0; j < 16; ++j) {
        int idx = lane + 32 * j;
        atomicAdd(out + idx, v[idx] * w[j >> 1]);
    }
}

__global__ void gelu_kernel(float* x, long n)
{
    long i = blockIdx.x * (long)blockDim.x + threadIdx.x;
    if (i >= n) return;
    float v = x[i];
    x[i] = 0.5f * v * (1.f + erff(v * 0.70710678118f));
}

__global__ void skip_kernel(float* out, const float* xin, const float* skip, long n)
{
    long i = blockIdx.x * (long)blockDim.x + threadIdx.x;
    if (i >= n) return;
    float a = 1.f / (1.f + __expf(-skip[0]));
    out[i] = a * out[i] + (1.f - a) * xin[i];
}

// ---------------- launch ----------------
extern "C" void kernel_launch(void* const* d_in, const int* in_sizes, int n_in,
                              void* d_out, int out_size, void* d_ws, size_t ws_size,
                              hipStream_t stream)
{
    const float* x     = (const float*)d_in[0];
    const int*   eidx  = (const int*)  d_in[1];
    const float* eattr = (const float*)d_in[2];
    const float* Wkqv0 = (const float*)d_in[3];
    const float* bkqv0 = (const float*)d_in[4];
    const float* Wk0   = (const float*)d_in[5];
    const float* Wv0   = (const float*)d_in[6];
    const float* prel0 = (const float*)d_in[7];
    const float* Wout0 = (const float*)d_in[8];
    const float* bout0 = (const float*)d_in[9];
    const float* Wkqv1 = (const float*)d_in[10];
    const float* bkqv1 = (const float*)d_in[11];
    const float* Wk1   = (const float*)d_in[12];
    const float* Wv1   = (const float*)d_in[13];
    const float* prel1 = (const float*)d_in[14];
    const float* Wout1 = (const float*)d_in[15];
    const float* bout1 = (const float*)d_in[16];
    const float* skip1 = (const float*)d_in[17];
    const float* Wfc   = (const float*)d_in[18];
    const float* bfc   = (const float*)d_in[19];
    const float* We1   = (const float*)d_in[20];
    const float* be1   = (const float*)d_in[21];
    const float* We2   = (const float*)d_in[22];
    const float* be2   = (const float*)d_in[23];
    float* out = (float*)d_out;

    // workspace layout (floats)
    float* ws = (float*)d_ws;
    size_t off = 0;
    float* kqv   = ws + off; off += (size_t)NN * 3 * HIDC;
    float* krel  = ws + off; off += (size_t)RR * NN * HIDC;
    float* vrel  = ws + off; off += (size_t)RR * NN * HIDC;
    float* alpha = ws + off; off += (size_t)RR * EE * NHEAD;
    float* amax  = ws + off; off += (size_t)NN * NHEAD;
    float* denom = ws + off; off += (size_t)NN * NHEAD;
    float* agg   = ws + off; off += (size_t)NN * HIDC;
    float* h1    = ws + off; off += (size_t)NN * HIDC;
    float* h2    = ws + off; off += (size_t)NN * HIDC;
    float* ehid  = ws + off; off += (size_t)RR * EE * HIDC;

    const dim3 blk(256);
    const int gM  = (NN + TM - 1) / TM;          // node-row blocks
    const int gMe = (RR * EE + TM - 1) / TM;     // edge-row blocks

    auto run_layer = [&](const float* xin, int cin,
                         const float* Wkqv, const float* bkqv,
                         const float* Wk, const float* Wv, const float* prel,
                         const float* Wout, const float* bout, float* hout) {
        // kqv = xin @ Wkqv + bkqv
        gemm_kernel<<<dim3(gM, (3 * HIDC) / TN, 1), blk, 0, stream>>>(
            xin, cin, Wkqv, 3 * HIDC, kqv, 3 * HIDC, bkqv,
            NN, 3 * HIDC, cin, 0);
        // per-(r,h) relation transforms of k and v
        gemm_rel_kernel<<<dim3(gM, DH / TN, RR * NHEAD), blk, 0, stream>>>(
            kqv, 0, Wk, krel);
        gemm_rel_kernel<<<dim3(gM, DH / TN, RR * NHEAD), blk, 0, stream>>>(
            kqv, 2 * HIDC, Wv, vrel);
        // attention
        init_attn<<<((long)NN * HIDC + 255) / 256, blk, 0, stream>>>(amax, denom, agg);
        attn_alpha<<<(RR * EE + 7) / 8, blk, 0, stream>>>(kqv, krel, eidx, prel, alpha, amax);
        attn_exp<<<((long)RR * EE * NHEAD + 255) / 256, blk, 0, stream>>>(alpha, eidx, amax, denom);
        attn_scatter<<<(RR * EE + 7) / 8, blk, 0, stream>>>(alpha, denom, vrel, eidx, agg);
        gelu_kernel<<<((long)NN * HIDC + 255) / 256, blk, 0, stream>>>(agg, (long)NN * HIDC);
        // hout = gelu(agg) @ Wout + bout
        gemm_kernel<<<dim3(gM, HIDC / TN, 1), blk, 0, stream>>>(
            agg, HIDC, Wout, HIDC, hout, HIDC, bout, NN, HIDC, HIDC, 0);
    };

    run_layer(x,  CIN,  Wkqv0, bkqv0, Wk0, Wv0, prel0, Wout0, bout0, h1);
    run_layer(h1, HIDC, Wkqv1, bkqv1, Wk1, Wv1, prel1, Wout1, bout1, h2);
    skip_kernel<<<((long)NN * HIDC + 255) / 256, blk, 0, stream>>>(
        h2, h1, skip1, (long)NN * HIDC);

    // node_embeds = h2 @ Wfc + bfc  -> d_out[0 : NN*OUTC)
    gemm_kernel<<<dim3(gM, OUTC / TN, 1), blk, 0, stream>>>(
        h2, HIDC, Wfc, OUTC, out, OUTC, bfc, NN, OUTC, HIDC, 0);

    // edge_embeds = relu(e @ We1 + be1) @ We2 + be2 -> d_out[NN*OUTC : ...)
    gemm_kernel<<<dim3(gMe, HIDC / TN, 1), blk, 0, stream>>>(
        eattr, CEDGE, We1, HIDC, ehid, HIDC, be1,
        RR * EE, HIDC, CEDGE, 1);
    gemm_kernel<<<dim3(gMe, OUTC / TN, 1), blk, 0, stream>>>(
        ehid, HIDC, We2, OUTC, out + (size_t)NN * OUTC, OUTC, be2,
        RR * EE, OUTC, HIDC, 0);
}
